// MotifT_13486197309991
// MI455X (gfx1250) — matrix operand order repaired and checked by reading, not compiled, not run on hardware
//
#include <hip/hip_runtime.h>

// ---- model config (matches reference) ----
#define BB   2
#define TT   1024
#define BT   2048      // B*T
#define DD   1024
#define HH   16
#define CHD  32        // q/k head dim
#define MHD  64        // v head dim
#define DE   4096
#define DG   1024
#define LLAY 6
#define VV   32000

typedef __attribute__((ext_vector_type(16))) __bf16 v16bf;
typedef __attribute__((ext_vector_type(8)))  float  v8f;
typedef __attribute__((ext_vector_type(4)))  unsigned int v4u;
typedef __attribute__((ext_vector_type(8)))  int v8i_t;
typedef __attribute__((ext_vector_type(4)))  int v4i_t;

union Frag {
  v16bf v;
  unsigned short s[16];
  uint4 q[2];
};

__device__ __forceinline__ unsigned short f2b(float f) {
  unsigned int u = __float_as_uint(f);
  u += 0x7FFFu + ((u >> 16) & 1u);       // round-to-nearest-even
  return (unsigned short)(u >> 16);
}
__device__ __forceinline__ float sigf(float x) {
  return __builtin_amdgcn_rcpf(1.0f + __expf(-x));   // v_exp + v_rcp
}
__device__ __forceinline__ float siluf(float x) { return x * sigf(x); }

__device__ __forceinline__ float hmax16(float v) {
  v = fmaxf(v, __shfl_xor(v, 1));
  v = fmaxf(v, __shfl_xor(v, 2));
  v = fmaxf(v, __shfl_xor(v, 4));
  v = fmaxf(v, __shfl_xor(v, 8));
  return v;
}
__device__ __forceinline__ float hsum16(float v) {
  v += __shfl_xor(v, 1);
  v += __shfl_xor(v, 2);
  v += __shfl_xor(v, 4);
  v += __shfl_xor(v, 8);
  return v;
}

// ---------------- embedding: x = te[ids] + pe ----------------
__global__ void __launch_bounds__(256) embed_kernel(
    const int* __restrict__ ids, const float* __restrict__ te,
    const float* __restrict__ pe, float* __restrict__ x) {
  int row = blockIdx.x;                 // 0..BT-1
  int t = row % TT;
  int id = ids[row];
  for (int i = threadIdx.x; i < DD; i += 256)
    x[(size_t)row * DD + i] = te[(size_t)id * DD + i] + pe[(size_t)t * DD + i];
}

// ---------------- layernorm -> bf16 ----------------
__global__ void __launch_bounds__(256) ln_kernel(
    const float* __restrict__ x, const float* __restrict__ g,
    const float* __restrict__ b, unsigned short* __restrict__ out) {
  int row = blockIdx.x;
  int tid = threadIdx.x;
  const float* xr = x + (size_t)row * DD;
  __shared__ float red[8];

  float s = 0.0f;
  for (int i = tid; i < DD; i += 256) s += xr[i];
#pragma unroll
  for (int m = 1; m < 32; m <<= 1) s += __shfl_xor(s, m);
  if ((tid & 31) == 0) red[tid >> 5] = s;
  __syncthreads();
  float mean = 0.0f;
#pragma unroll
  for (int i = 0; i < 8; i++) mean += red[i];
  mean *= (1.0f / DD);
  __syncthreads();

  float vs = 0.0f;
  for (int i = tid; i < DD; i += 256) {
    float d = xr[i] - mean;
    vs += d * d;
  }
#pragma unroll
  for (int m = 1; m < 32; m <<= 1) vs += __shfl_xor(vs, m);
  if ((tid & 31) == 0) red[tid >> 5] = vs;
  __syncthreads();
  float var = 0.0f;
#pragma unroll
  for (int i = 0; i < 8; i++) var += red[i];
  var *= (1.0f / DD);
  float rstd = rsqrtf(var + 1e-5f);

  for (int i = tid; i < DD; i += 256)
    out[(size_t)row * DD + i] = f2b((xr[i] - mean) * rstd * g[i] + b[i]);
}

// ---------------- elementwise u = silu(e*g) -> bf16 ----------------
__global__ void __launch_bounds__(256) mulsilu_f2b_kernel(
    const float* __restrict__ a, const float* __restrict__ b,
    unsigned short* __restrict__ out, int n) {
  int i = blockIdx.x * 256 + threadIdx.x;
  if (i < n) out[i] = f2b(siluf(a[i] * b[i]));
}

// ---------------- weight pack: f32 [K,N] -> bf16 B-fragment order ----------------
// P[kt][nt][lane][e], e contiguous:  k = kt*32 + 16*(lane/16) + e ; n = nt*16 + lane%16
// trans==1 reads W as [N, ldsrc] (i.e. B[k][n] = W[n][k]) -- for the tied lm head.
__global__ void __launch_bounds__(256) pack_b_kernel(
    const float* __restrict__ W, unsigned short* __restrict__ P,
    int K, int N, int trans, int ldsrc) {
  int idx = blockIdx.x * 256 + threadIdx.x;
  if (idx >= K * N) return;
  int e = idx & 15;
  int lane = (idx >> 4) & 31;
  int chunk = idx >> 9;                 // kt*(N/16) + nt
  int ntiles = N >> 4;
  int kt = chunk / ntiles, nt = chunk % ntiles;
  int g = lane >> 4, l16 = lane & 15;
  int k = kt * 32 + 16 * g + e;
  int n = nt * 16 + l16;
  float v = trans ? W[(size_t)n * ldsrc + k] : W[(size_t)k * ldsrc + n];
  P[idx] = f2b(v);
}

// ---------------- generic bf16 WMMA GEMM, packed B, TDM-staged panels ----------------
// out = epilogue( A[M,K]bf16 * B[K,N]bf16(packed) + bias[N] )
// act: 0=none 1=silu 2=sigmoid
// Cb != null  -> bf16 store to Cb ; else res==1 -> C += v ; else C = v
// Block: 128 thr = 4 waves; block tile 64(M) x 128(N); wave tile 64 x 32.
// Packed B panel (4 kt x 8 nt chunks = 32 KB) staged into LDS by the TDM.
#define KSTAGE 128
__global__ void __launch_bounds__(128) gemm_kernel(
    const unsigned short* __restrict__ A, const unsigned short* __restrict__ Bp,
    const float* __restrict__ bias, float* __restrict__ C,
    unsigned short* __restrict__ Cb,
    int M, int N, int K, int act, int residual) {
  __shared__ unsigned short Bsh[KSTAGE * 128];   // 32 KB = 4 kt-rows x 4096 elements

  int wave = threadIdx.x >> 5;
  int lane = threadIdx.x & 31;
  int g = lane >> 4, l16 = lane & 15;
  int tilesN = N >> 7;                           // 128-col block tiles
  int tM = blockIdx.x / tilesN, tN = blockIdx.x % tilesN;
  int row0 = tM * 64;
  int colbase = tN * 128;
  int ntiles = N >> 4;                           // nt chunks per kt row in packed B

  v8f acc[8];                                    // [mb(4)][j(2)]
#pragma unroll
  for (int a = 0; a < 8; a++)
#pragma unroll
    for (int r = 0; r < 8; r++) acc[a][r] = 0.0f;

  for (int ks = 0; ks < K; ks += KSTAGE) {
    __syncthreads();                             // all waves done with Bsh
    if (wave == 0) {
      // ---- TDM 2D copy of packed panel: 4 kt-rows of 4096 contiguous elements ----
      unsigned lds_off = (unsigned)(((size_t)(void*)Bsh) & 0xFFFFFFFFu);
      unsigned long long ga = (unsigned long long)(const void*)
          (Bp + ((size_t)(ks >> 5) * ntiles + (colbase >> 4)) * 512);
      v4u g0;
      g0[0] = 1u;                                // count=1, user descriptor
      g0[1] = lds_off;                           // lds_addr
      g0[2] = (unsigned)(ga & 0xFFFFFFFFu);      // global_addr[31:0]
      g0[3] = (unsigned)((ga >> 32) & 0x01FFFFFFu) | (2u << 30);  // addr hi | type=2
      v8i_t g1;
      g1[0] = (int)(1u << 16);                   // workgroup_mask=0, data_size=1 (2B)
      g1[1] = (int)(4096u << 16);                // tensor_dim0 = 4096 (tile-local)
      g1[2] = (int)(4u << 16);                   // tensor_dim1 = 4
      g1[3] = (int)(4096u << 16);                // tile_dim0 = 4096 elements
      g1[4] = (int)4;                            // tile_dim1 = 4, tile_dim2 = 0
      g1[5] = (int)(ntiles * 512);               // tensor_dim0_stride = N*32 elements
      g1[6] = 0;
      g1[7] = 0;
      v4i_t gz;
      gz[0] = 0; gz[1] = 0; gz[2] = 0; gz[3] = 0;
#if defined(__clang_major__) && __clang_major__ >= 23
      v8i_t gz8;
#pragma unroll
      for (int i = 0; i < 8; i++) gz8[i] = 0;
      __builtin_amdgcn_tensor_load_to_lds(g0, g1, gz, gz, gz8, 0);
#else
      __builtin_amdgcn_tensor_load_to_lds(g0, g1, gz, gz, 0);
#endif
      __builtin_amdgcn_s_wait_tensorcnt(0);
    }
    __syncthreads();                             // B panel visible to all waves

#pragma unroll
    for (int kk = 0; kk < KSTAGE; kk += 32) {
      // two B fragments, each = 32 contiguous bytes per lane in the packed panel
      const unsigned short* bp =
          Bsh + ((size_t)((kk >> 5) * 8 + wave * 2) * 512) + lane * 16;
      Frag bf[2];
      bf[0].q[0] = *(const uint4*)(bp);
      bf[0].q[1] = *(const uint4*)(bp + 8);
      bf[1].q[0] = *(const uint4*)(bp + 512);
      bf[1].q[1] = *(const uint4*)(bp + 520);
#pragma unroll
      for (int mb = 0; mb < 4; mb++) {
        // A fragment: lane row = row0+mb*16+l16 ; e -> k+16g+e
        Frag af;
        const unsigned short* ap =
            A + (size_t)(row0 + mb * 16 + l16) * K + ks + kk + 16 * g;
        af.q[0] = *(const uint4*)(ap);
        af.q[1] = *(const uint4*)(ap + 8);
#pragma unroll
        for (int j = 0; j < 2; j++)
          acc[mb * 2 + j] = __builtin_amdgcn_wmma_f32_16x16x32_bf16(
              false, af.v, false, bf[j].v, (short)0, acc[mb * 2 + j], false, false);
      }
    }
  }

  float bv[2];
#pragma unroll
  for (int j = 0; j < 2; j++)
    bv[j] = bias ? bias[colbase + wave * 32 + j * 16 + l16] : 0.0f;
#pragma unroll
  for (int mb = 0; mb < 4; mb++) {
#pragma unroll
    for (int j = 0; j < 2; j++) {
      int col = colbase + wave * 32 + j * 16 + l16;
#pragma unroll
      for (int r = 0; r < 8; r++) {
        int row = row0 + mb * 16 + r + 8 * g;
        float v = acc[mb * 2 + j][r] + bv[j];
        if (act == 1) v = siluf(v);
        else if (act == 2) v = sigf(v);
        size_t idx = (size_t)row * N + col;
        if (Cb) Cb[idx] = f2b(v);
        else if (residual) C[idx] += v;
        else C[idx] = v;
      }
    }
  }
}

// ---------------- flash attention (bf16 q/k/v in, bf16 y out) ----------------
// one wave per (b, h, 16-row q tile). CHD=32 => one WMMA per S tile.
__global__ void __launch_bounds__(32) attn_kernel(
    const unsigned short* __restrict__ q, const unsigned short* __restrict__ k,
    const unsigned short* __restrict__ v, unsigned short* __restrict__ y) {
  const int qt = blockIdx.x % (TT / 16);
  const int h = (blockIdx.x / (TT / 16)) % HH;
  const int b = blockIdx.x / ((TT / 16) * HH);
  const int lane = threadIdx.x;
  const int g = lane >> 4, l16 = lane & 15;
  const int q0 = qt * 16;
  const float scale = 0.17677669529663687f;  // 1/sqrt(32)

  __shared__ unsigned short P[256];          // 16x16 P tile (bf16 bits)

  // Q fragment (A layout): two contiguous 128-bit loads, no conversion
  Frag Qa;
  {
    const unsigned short* qr =
        q + (size_t)(b * TT + q0 + l16) * (HH * CHD) + h * CHD;
    Qa.q[0] = *(const uint4*)(qr + 16 * g);       // e0..7  -> kd 16g..16g+7
    Qa.q[1] = *(const uint4*)(qr + 16 * g + 8);   // e8..15 -> kd 16g+8..
  }

  v8f o[4];
  float mrow[8], lrow[8];
#pragma unroll
  for (int j = 0; j < 4; j++)
#pragma unroll
    for (int r = 0; r < 8; r++) o[j][r] = 0.0f;
#pragma unroll
  for (int r = 0; r < 8; r++) { mrow[r] = -1e30f; lrow[r] = 0.0f; }

  for (int kt = 0; kt <= qt; kt++) {
    // K^T fragment (B layout): col = key pos kt*16+l16, k-dim = 16*g + e (contiguous)
    Frag Kf;
    {
      const unsigned short* kr =
          k + (size_t)(b * TT + kt * 16 + l16) * (HH * CHD) + h * CHD + 16 * g;
      Kf.q[0] = *(const uint4*)(kr);
      Kf.q[1] = *(const uint4*)(kr + 8);
    }

    v8f z;
#pragma unroll
    for (int r = 0; r < 8; r++) z[r] = 0.0f;
    v8f S = __builtin_amdgcn_wmma_f32_16x16x32_bf16(
        false, Qa.v, false, Kf.v, (short)0, z, false, false);

#pragma unroll
    for (int r = 0; r < 8; r++) {
      int rowt = q0 + r + 8 * g;
      int colt = kt * 16 + l16;
      float sv = S[r] * scale;
      if (colt > rowt) sv = -1e30f;         // causal mask
      float rmax = hmax16(sv);
      float mnew = fmaxf(mrow[r], rmax);
      float alpha = __expf(mrow[r] - mnew);
      float pv = __expf(sv - mnew);
      float psum = hsum16(pv);
      lrow[r] = lrow[r] * alpha + psum;
      mrow[r] = mnew;
#pragma unroll
      for (int j = 0; j < 4; j++) o[j][r] *= alpha;
      P[(r + 8 * g) * 16 + l16] = f2b(pv);  // P[query][key]
    }
    __syncthreads();

    // P as A fragment: lane row = l16 (query), k = 16*g + e (only g==0 are real keys)
    Frag Pa;
#pragma unroll
    for (int e = 0; e < 16; e++)
      Pa.s[e] = (g == 0) ? P[l16 * 16 + e] : (unsigned short)0;

#pragma unroll
    for (int j = 0; j < 4; j++) {
      // V as B fragment: krow = 16*g + e (only g==0 rows are real keys), col = v-dim
      Frag Vf;
#pragma unroll
      for (int e = 0; e < 16; e++)
        Vf.s[e] = (g == 0)
            ? v[(size_t)(b * TT + kt * 16 + e) * (HH * MHD) + h * MHD + j * 16 + l16]
            : (unsigned short)0;
      o[j] = __builtin_amdgcn_wmma_f32_16x16x32_bf16(
          false, Pa.v, false, Vf.v, (short)0, o[j], false, false);
    }
    __syncthreads();
  }

#pragma unroll
  for (int r = 0; r < 8; r++) {
    float inv = 1.0f / lrow[r];
    int rowt = b * TT + q0 + r + 8 * g;
#pragma unroll
    for (int j = 0; j < 4; j++)
      y[(size_t)rowt * (HH * MHD) + h * MHD + j * 16 + l16] = f2b(o[j][r] * inv);
  }
}

// ---------------- host orchestration ----------------
extern "C" void kernel_launch(void* const* d_in, const int* in_sizes, int n_in,
                              void* d_out, int out_size, void* d_ws, size_t ws_size,
                              hipStream_t stream) {
  (void)in_sizes; (void)n_in; (void)out_size; (void)ws_size;
  const int*   ids    = (const int*)d_in[0];
  const float* te     = (const float*)d_in[1];
  const float* pe     = (const float*)d_in[2];
  const float* ln1_g  = (const float*)d_in[3];
  const float* ln1_b  = (const float*)d_in[4];
  const float* qp_w   = (const float*)d_in[5];
  const float* qp_b   = (const float*)d_in[6];
  const float* kp_w   = (const float*)d_in[7];
  const float* kp_b   = (const float*)d_in[8];
  const float* vp_w   = (const float*)d_in[9];
  const float* vp_b   = (const float*)d_in[10];
  const float* out_w  = (const float*)d_in[11];
  const float* out_b  = (const float*)d_in[12];
  const float* ln2_g  = (const float*)d_in[13];
  const float* ln2_b  = (const float*)d_in[14];
  const float* exp_w  = (const float*)d_in[15];
  const float* exp_b  = (const float*)d_in[16];
  const float* gate_w = (const float*)d_in[17];
  const float* gate_b = (const float*)d_in[18];
  const float* gup_w  = (const float*)d_in[19];
  const float* gup_b  = (const float*)d_in[20];
  const float* comp_w = (const float*)d_in[21];
  const float* comp_b = (const float*)d_in[22];
  const float* lnf_g  = (const float*)d_in[23];
  const float* lnf_b  = (const float*)d_in[24];
  float* logits = (float*)d_out;

  // workspace carve (all 256B aligned)
  char* p = (char*)d_ws;
  auto take = [&](size_t bytes) -> char* {
    char* r = p;
    p += (bytes + 255) & ~(size_t)255;
    return r;
  };
  float*          x    = (float*)take((size_t)BT * DD * 4);
  unsigned short* hbf  = (unsigned short*)take((size_t)BT * DD * 2);
  unsigned short* qbf  = (unsigned short*)take((size_t)BT * HH * CHD * 2);
  unsigned short* kbf  = (unsigned short*)take((size_t)BT * HH * CHD * 2);
  unsigned short* vbf  = (unsigned short*)take((size_t)BT * HH * MHD * 2);
  unsigned short* ybf  = (unsigned short*)take((size_t)BT * HH * MHD * 2);
  float*          ebuf = (float*)take((size_t)BT * DE * 4);
  float*          gbuf = (float*)take((size_t)BT * DE * 4);
  unsigned short* tbf  = (unsigned short*)take((size_t)BT * DG * 2);
  unsigned short* ubf  = (unsigned short*)take((size_t)BT * DE * 2);
  unsigned short* wpb  = (unsigned short*)take((size_t)DE * DD * 2);  // packed weight
  unsigned short* tep  = (unsigned short*)take((size_t)DD * VV * 2);  // packed lm head

  auto pack = [&](const float* W, unsigned short* P, int K, int N, int trans, int ld) {
    int total = K * N;
    pack_b_kernel<<<(total + 255) / 256, 256, 0, stream>>>(W, P, K, N, trans, ld);
  };
  // f32 output (optionally residual accumulate)
  auto gemm_f = [&](const unsigned short* A, const unsigned short* Bp, const float* bias,
                    float* C, int M, int N, int K, int act, int res) {
    int blocks = (M / 64) * (N / 128);
    gemm_kernel<<<blocks, 128, 0, stream>>>(A, Bp, bias, C, nullptr, M, N, K, act, res);
  };
  // bf16 output
  auto gemm_b = [&](const unsigned short* A, const unsigned short* Bp, const float* bias,
                    unsigned short* Cb, int M, int N, int K, int act) {
    int blocks = (M / 64) * (N / 128);
    gemm_kernel<<<blocks, 128, 0, stream>>>(A, Bp, bias, nullptr, Cb, M, N, K, act, 0);
  };

  embed_kernel<<<BT, 256, 0, stream>>>(ids, te, pe, x);

  for (int l = 0; l < LLAY; l++) {
    // --- attention sublayer ---
    ln_kernel<<<BT, 256, 0, stream>>>(x, ln1_g + (size_t)l * DD, ln1_b + (size_t)l * DD, hbf);

    pack(qp_w + (size_t)l * DD * HH * CHD, wpb, DD, HH * CHD, 0, HH * CHD);
    gemm_b(hbf, wpb, qp_b + (size_t)l * HH * CHD, qbf, BT, HH * CHD, DD, 0);

    pack(kp_w + (size_t)l * DD * HH * CHD, wpb, DD, HH * CHD, 0, HH * CHD);
    gemm_b(hbf, wpb, kp_b + (size_t)l * HH * CHD, kbf, BT, HH * CHD, DD, 0);

    pack(vp_w + (size_t)l * DD * HH * MHD, wpb, DD, HH * MHD, 0, HH * MHD);
    gemm_b(hbf, wpb, vp_b + (size_t)l * HH * MHD, vbf, BT, HH * MHD, DD, 0);

    attn_kernel<<<BB * HH * (TT / 16), 32, 0, stream>>>(qbf, kbf, vbf, ybf);

    pack(out_w + (size_t)l * HH * MHD * DD, wpb, HH * MHD, DD, 0, DD);
    gemm_f(ybf, wpb, out_b + (size_t)l * DD, x, BT, DD, HH * MHD, 0, 1);  // residual

    // --- gated FFN sublayer ---
    ln_kernel<<<BT, 256, 0, stream>>>(x, ln2_g + (size_t)l * DD, ln2_b + (size_t)l * DD, hbf);

    pack(exp_w + (size_t)l * DD * DE, wpb, DD, DE, 0, DE);
    gemm_f(hbf, wpb, exp_b + (size_t)l * DE, ebuf, BT, DE, DD, 0, 0);

    pack(gate_w + (size_t)l * DD * DG, wpb, DD, DG, 0, DG);
    gemm_b(hbf, wpb, gate_b + (size_t)l * DG, tbf, BT, DG, DD, 1);     // fused silu -> bf16

    pack(gup_w + (size_t)l * DG * DE, wpb, DG, DE, 0, DE);
    gemm_f(tbf, wpb, gup_b + (size_t)l * DE, gbuf, BT, DE, DG, 2, 0);  // fused sigmoid

    mulsilu_f2b_kernel<<<(BT * DE + 255) / 256, 256, 0, stream>>>(ebuf, gbuf, ubf, BT * DE);

    pack(comp_w + (size_t)l * DE * DD, wpb, DE, DD, 0, DD);
    gemm_f(ubf, wpb, comp_b + (size_t)l * DD, x, BT, DD, DE, 0, 1);    // residual
  }

  // --- final LN + tied lm head (packed transpose of te) ---
  ln_kernel<<<BT, 256, 0, stream>>>(x, lnf_g, lnf_b, hbf);
  pack(te, tep, DD, VV, 1, DD);
  gemm_f(hbf, tep, nullptr, logits, BT, VV, DD, 0, 0);
}
